// GraphSAGEMean_41540923687233
// MI455X (gfx1250) — matrix operand order repaired and checked by reading, not compiled, not run on hardware
//
#include <hip/hip_runtime.h>

typedef __attribute__((ext_vector_type(16))) __bf16 bf16x16;
typedef __attribute__((ext_vector_type(8)))  __bf16 bf16x8;
typedef __attribute__((ext_vector_type(8)))  float  f32x8;

#define PITCH       264    // 256 + 8 bf16 pad -> conflict-minimal LDS rows
#define RPB         128    // rows per block
#define HALF_ELEMS  32768  // bf16 elems per 64KB weight stage

// ---------------------------------------------------------------------------
// Pack f32 weight W[k, nout] (row-major, fan_in=256) into bf16 fragments for
// v_wmma_f32_16x16x32_bf16, K-half major so each 64KB stage = one K-half with
// ALL n-tiles:  f = h*(ntc*4) + nt*4 + ktl   (h = kt>>2, ktl = kt&3)
//   lane L: col = nt*16 + (L&15); 16 bf16 = K kt*32 + (L>=16)*8 + {0..7,16..23}
// ---------------------------------------------------------------------------
__global__ __launch_bounds__(256) void pack_w(const float* __restrict__ W,
                                              __bf16* __restrict__ dst,
                                              int nout) {
  int t = blockIdx.x * 256 + threadIdx.x;
  if (t >= nout * 256) return;
  int s    = t & 15;
  int L    = (t >> 4) & 31;
  int f    = t >> 9;
  int ntc4 = (nout >> 4) << 2;          // (nout/16)*4
  int h    = f / ntc4;
  int rr   = f - h * ntc4;
  int nt   = rr >> 2;
  int ktl  = rr & 3;
  int kt   = h * 4 + ktl;
  int col  = nt * 16 + (L & 15);
  int k    = kt * 32 + ((L >> 4) << 3) + (s & 7) + ((s >> 3) << 4);
  dst[t] = (__bf16)W[k * nout + col];
}

// ---- async global -> LDS staging (gfx1250, ASYNCcnt-tracked) ----
__device__ __forceinline__ void stage_async(const __bf16* __restrict__ src,
                                            __bf16* dstLds, int tid) {
  unsigned ldsBase = (unsigned)(unsigned long long)(void*)dstLds;
  const char* g = (const char*)src;
#pragma unroll
  for (int j = 0; j < 16; ++j) {          // 256 thr x 16 x 16B = 64KB
    unsigned off = (unsigned)(tid + j * 256) * 16u;
    unsigned long long ga = (unsigned long long)(g + off);
    unsigned la = ldsBase + off;
    asm volatile("global_load_async_to_lds_b128 %0, %1, off"
                 :: "v"(la), "v"(ga) : "memory");
  }
}
__device__ __forceinline__ void wait_async_all() {
  asm volatile("s_wait_asynccnt 0x0" ::: "memory");
}

// A fragment (16x32 bf16): lanes 0-15 row=lane, K {0..7,16..23}; lanes 16-31
// row=lane-16, K {8..15,24..31}. aBase already includes (row0 + lane&15)*PITCH.
__device__ __forceinline__ bf16x16 load_a(const __bf16* aBase, int kt, int lane) {
  int kbase = kt * 32 + ((lane & 16) ? 8 : 0);
  const bf16x8* p = (const bf16x8*)(aBase + kbase);
  bf16x8 lo = p[0];
  bf16x8 hi = p[2];
  bf16x16 a;
#pragma unroll
  for (int i = 0; i < 8; ++i) { a[i] = lo[i]; a[i + 8] = hi[i]; }
  return a;
}

// MS row-strips x 4 n-tiles over one K-half (4 kt). bb = wbuf + lane*16 (one
// base VGPR; all fragment offsets are constants < 64KB -> fold into ds offset).
template <int MS>
__device__ __forceinline__ void gemm_tiles(const __bf16* aBase, const __bf16* bb,
                                           int kt0, int nt0, int fAdd, int lane,
                                           f32x8* acc) {
#pragma unroll
  for (int ktl = 0; ktl < 4; ++ktl) {
    bf16x16 a[MS];
#pragma unroll
    for (int ms = 0; ms < MS; ++ms)
      a[ms] = load_a(aBase + ms * 16 * PITCH, kt0 + ktl, lane);
#pragma unroll
    for (int ntl = 0; ntl < 4; ++ntl) {
      int fl = fAdd + (nt0 + ntl) * 4 + ktl;
      bf16x16 b = *(const bf16x16*)(bb + fl * 512);
#pragma unroll
      for (int ms = 0; ms < MS; ++ms)
        acc[ms * 4 + ntl] = __builtin_amdgcn_wmma_f32_16x16x32_bf16(
            false, a[ms], false, b, (short)0, acc[ms * 4 + ntl], false, false);
    }
  }
}

// bias+relu+bf16 writeback: C frag row = r + 8*(lane/16), col = nt*16 + lane%16
__device__ __forceinline__ void writeback_relu(__bf16* act, int rows0, int nt0,
                                               int lane,
                                               const float* __restrict__ bias,
                                               const f32x8* acc) {
  int col0  = lane & 15;
  int rlane = (lane >> 4) << 3;
#pragma unroll
  for (int ms = 0; ms < 4; ++ms) {
#pragma unroll
    for (int ntl = 0; ntl < 4; ++ntl) {
      int nt = nt0 + ntl;
      float bn = bias[nt * 16 + col0];
#pragma unroll
      for (int r = 0; r < 8; ++r) {
        float v = acc[ms * 4 + ntl][r] + bn;
        v = v > 0.f ? v : 0.f;
        act[(rows0 + ms * 16 + rlane + r) * PITCH + nt * 16 + col0] = (__bf16)v;
      }
    }
  }
}

__global__ __launch_bounds__(256) void sage_fused(
    const float* __restrict__ E,
    const __bf16* __restrict__ w1, const float* __restrict__ b1,
    const __bf16* __restrict__ w2, const float* __restrict__ b2,
    const __bf16* __restrict__ w3, const float* __restrict__ b3,
    const __bf16* __restrict__ wo, const float* __restrict__ bo,
    float* __restrict__ out, float* __restrict__ agg, int n) {
  __shared__ __bf16 act[RPB * PITCH];     // 66 KB, LDS offset 0 (A offsets fold)
  __shared__ __bf16 wbufA[HALF_ELEMS];    // 64 KB ping
  __shared__ __bf16 wbufB[HALF_ELEMS];    // 64 KB pong

  const int  tid  = threadIdx.x;
  const int  lane = tid & 31;
  const int  wave = tid >> 5;
  const long blk0 = (long)blockIdx.x * RPB;

  stage_async(w1, wbufA, tid);            // step0 overlaps the E read

  // ---- Phase 1: E rows (f32) -> fused row-mean + bf16 into LDS ----
  {
    const long rowBase = blk0 + wave * 16;
    for (int r = 0; r < 16; ++r) {
      long row = rowBase + r;
      float4 a0 = make_float4(0.f, 0.f, 0.f, 0.f);
      float4 a1 = make_float4(0.f, 0.f, 0.f, 0.f);
      if (row < n) {
        const float4* p = (const float4*)(E + row * 256 + lane * 8);
        a0 = p[0]; a1 = p[1];
      }
      float s = a0.x + a0.y + a0.z + a0.w + a1.x + a1.y + a1.z + a1.w;
#pragma unroll
      for (int off = 16; off > 0; off >>= 1) s += __shfl_xor(s, off, 32);
      if (lane == 0 && row < n) agg[row] = s * (1.0f / 256.0f);

      bf16x8 bv;
      bv[0] = (__bf16)a0.x; bv[1] = (__bf16)a0.y; bv[2] = (__bf16)a0.z; bv[3] = (__bf16)a0.w;
      bv[4] = (__bf16)a1.x; bv[5] = (__bf16)a1.y; bv[6] = (__bf16)a1.z; bv[7] = (__bf16)a1.w;
      *(bf16x8*)(&act[(wave * 16 + r) * PITCH + lane * 8]) = bv;
    }
  }

  // compute-tile mapping: 64x64 wave tiles (2 row-strips x 4 col-strips)
  const int rs  = wave >> 2;              // 0..1  -> rows rs*64..+63
  const int cs  = wave & 3;               // 0..3  -> n-tiles cs*4..+3
  const __bf16* aBase = act + (rs * 64 + (lane & 15)) * PITCH;
  const __bf16* bbA   = wbufA + lane * 16;
  const __bf16* bbB   = wbufB + lane * 16;

  const f32x8 zero = {0.f, 0.f, 0.f, 0.f, 0.f, 0.f, 0.f, 0.f};
  f32x8 acc[16];

#define SYNC_ISSUE(SRC, DST)                         \
  do { wait_async_all(); __syncthreads();            \
       stage_async((SRC), (DST), tid); } while (0)
#define SYNC_LAST()                                  \
  do { wait_async_all(); __syncthreads(); } while (0)

  // ---------------- layer 1 ----------------
#pragma unroll
  for (int i = 0; i < 16; ++i) acc[i] = zero;
  SYNC_ISSUE(w1 + HALF_ELEMS, wbufB);               // W1 K-half1
  gemm_tiles<4>(aBase, bbA, 0, cs * 4, 0, lane, acc);
  SYNC_ISSUE(w2, wbufA);                            // W2 K-half0
  gemm_tiles<4>(aBase, bbB, 4, cs * 4, 0, lane, acc);
  __syncthreads();                                  // all readers done
  writeback_relu(act, rs * 64, cs * 4, lane, b1, acc);

  // ---------------- layer 2 ----------------
#pragma unroll
  for (int i = 0; i < 16; ++i) acc[i] = zero;
  SYNC_ISSUE(w2 + HALF_ELEMS, wbufB);               // W2 K-half1
  gemm_tiles<4>(aBase, bbA, 0, cs * 4, 0, lane, acc);
  SYNC_ISSUE(w3, wbufA);                            // W3 K-half0
  gemm_tiles<4>(aBase, bbB, 4, cs * 4, 0, lane, acc);
  __syncthreads();
  writeback_relu(act, rs * 64, cs * 4, lane, b2, acc);

  // ---------------- layer 3 ----------------
#pragma unroll
  for (int i = 0; i < 16; ++i) acc[i] = zero;
  SYNC_ISSUE(w3 + HALF_ELEMS, wbufB);               // W3 K-half1
  gemm_tiles<4>(aBase, bbA, 0, cs * 4, 0, lane, acc);
  SYNC_ISSUE(wo, wbufA);                            // Wo (full 64KB)
  gemm_tiles<4>(aBase, bbB, 4, cs * 4, 0, lane, acc);
  __syncthreads();
  writeback_relu(act, rs * 64, cs * 4, lane, b3, acc);

  // ------- output layer: 32x64 wave tiles (4 row-strips x 2 col-strips) -----
  {
    const int rs2 = wave >> 1;            // 0..3 -> rows rs2*32..+31
    const int cs2 = wave & 1;             // 0..1 -> n-tiles cs2*4..+3
    const __bf16* aBase2 = act + (rs2 * 32 + (lane & 15)) * PITCH;
#pragma unroll
    for (int i = 0; i < 8; ++i) acc[i] = zero;
    SYNC_LAST();
    gemm_tiles<2>(aBase2, bbA, 0, cs2 * 4, 0,  lane, acc);   // K-half0
    gemm_tiles<2>(aBase2, bbA, 4, cs2 * 4, 32, lane, acc);   // K-half1

    int  col0  = lane & 15;
    int  rlane = (lane >> 4) << 3;
#pragma unroll
    for (int ms = 0; ms < 2; ++ms) {
#pragma unroll
      for (int ntl = 0; ntl < 4; ++ntl) {
        int nt = cs2 * 4 + ntl;
        float bn = bo[nt * 16 + col0];
#pragma unroll
        for (int r = 0; r < 8; ++r) {
          long row = blk0 + rs2 * 32 + ms * 16 + rlane + r;
          if (row < n) out[row * 128 + nt * 16 + col0] = acc[ms * 4 + ntl][r] + bn;
        }
      }
    }
  }
#undef SYNC_ISSUE
#undef SYNC_LAST
}

// ---------------------------------------------------------------------------
// Inputs (setup_inputs order): 0=node_embeddings[N,256] f32, 1=adj_keys
// (mathematically unused: gather index == arange), 2..9 = W1,b1,W2,b2,W3,b3,
// Wo,bo (f32).  d_out = [ x (N*128 f32) | agg (N f32) ].  d_ws holds 448 KB
// of fragment-packed bf16 weights, rebuilt every call (deterministic).
// ---------------------------------------------------------------------------
extern "C" void kernel_launch(void* const* d_in, const int* in_sizes, int n_in,
                              void* d_out, int out_size, void* d_ws, size_t ws_size,
                              hipStream_t stream) {
  const float* E  = (const float*)d_in[0];
  const float* W1 = (const float*)d_in[2];
  const float* b1 = (const float*)d_in[3];
  const float* W2 = (const float*)d_in[4];
  const float* b2 = (const float*)d_in[5];
  const float* W3 = (const float*)d_in[6];
  const float* b3 = (const float*)d_in[7];
  const float* Wo = (const float*)d_in[8];
  const float* bo = (const float*)d_in[9];

  int n = in_sizes[0] / 256;   // N = 200000

  __bf16* w1p = (__bf16*)d_ws;
  __bf16* w2p = w1p + 256 * 256;
  __bf16* w3p = w2p + 256 * 256;
  __bf16* wop = w3p + 256 * 256;

  pack_w<<<256, 256, 0, stream>>>(W1, w1p, 256);
  pack_w<<<256, 256, 0, stream>>>(W2, w2p, 256);
  pack_w<<<256, 256, 0, stream>>>(W3, w3p, 256);
  pack_w<<<128, 256, 0, stream>>>(Wo, wop, 128);

  float* out = (float*)d_out;
  float* agg = out + (size_t)n * 128;
  int grid = (n + RPB - 1) / RPB;
  sage_fused<<<grid, 256, 0, stream>>>(E, w1p, b1, w2p, b2, w3p, b3, wop, bo,
                                       out, agg, n);
}
